// GWNetBlock_61675730370844
// MI455X (gfx1250) — matrix operand order repaired and checked by reading, not compiled, not run on hardware
//
#include <hip/hip_runtime.h>
#include <hip/hip_bf16.h>
#include <stdint.h>

// ---------------- CDNA5 WMMA types ----------------
typedef __attribute__((ext_vector_type(16))) __bf16 v16bf;
typedef __attribute__((ext_vector_type(8)))  __bf16 v8bf;
typedef __attribute__((ext_vector_type(8)))  float  v8f;

#define WMMA_BF16(A, Bm, Cm) \
    __builtin_amdgcn_wmma_f32_16x16x32_bf16(false, (A), false, (Bm), (short)0, (Cm), false, false)

static constexpr int Bc = 32, Cc = 32, Oc = 32, Nc = 512, Lc = 128;

// Branchless activations on the VALU fast path (v_exp_f32 / v_rcp_f32).
__device__ __forceinline__ float fast_sigmoid(float x) {
  x = fminf(fmaxf(x, -30.f), 30.f);
  return __builtin_amdgcn_rcpf(1.f + __builtin_amdgcn_exp2f(-1.442695041f * x));
}
__device__ __forceinline__ float fast_tanh(float x) {
  x = fminf(fmaxf(x, -15.f), 15.f);
  float e = __builtin_amdgcn_exp2f(-2.885390082f * x);
  return (1.f - e) * __builtin_amdgcn_rcpf(1.f + e);
}

// ---------------- fragment loaders ----------------
// A matrix 16x32 bf16: lane 0-15 row M=lane, K 0..15;
// lane 16-31 row M=lane-16, K 16..31 (matches swizzled B). Both 8-element groups are
// contiguous in a row-major [rows][ldm] array -> two aligned 16B loads.
union AFrag { v16bf v; struct { v8bf lo, hi; } p; };
__device__ __forceinline__ v16bf load_afrag(const __bf16* base, int ldm, int row0, int k0) {
  int lane = threadIdx.x & 31;
  int row  = row0 + (lane & 15);
  int half = lane >> 4;
  const __bf16* rp = base + row * ldm + k0 + half * 16;
  AFrag a;
  a.p.lo = *(const v8bf*)(rp);
  a.p.hi = *(const v8bf*)(rp + 8);
  return a.v;
}

// Fragment-ready swizzled B layout for a 32(K) x ncols operand:
//   element (k,c) lives at ((c>>4)*32 + ((k>>4)<<4) + (c&15))*16 + (k&15)
// so lane `ln` of column-tile t reads 16 contiguous bf16 at (t*32+ln)*16.
__device__ __forceinline__ int bswz(int k, int c) {
  return (((c >> 4) * 32) + ((k >> 4) << 4) + (c & 15)) * 16 + (k & 15);
}
__device__ __forceinline__ v16bf load_bfrag_sw(const __bf16* base, int t) {
  int lane = threadIdx.x & 31;
  return *(const v16bf*)(base + (t * 32 + lane) * 16);
}

// ---------------- k0: pack weights / supports to bf16, zero BN accumulators ----------------
__global__ void k0_prep(const float* __restrict__ s0, const float* __restrict__ s1,
                        const float* __restrict__ fw, const float* __restrict__ gw,
                        const float* __restrict__ g0w, const float* __restrict__ g1w,
                        const float* __restrict__ skw, const float* __restrict__ rsw,
                        __bf16* __restrict__ sup,    // [512][1024]
                        __bf16* __restrict__ wconv,  // [2][32][64]  f = k*32+c
                        __bf16* __restrict__ gcw,    // [2][32][32]
                        __bf16* __restrict__ swb,    // [32][32]
                        __bf16* __restrict__ rwb,    // [32][32]
                        float*  __restrict__ bn_acc) // [64]
{
  int idx = blockIdx.x * blockDim.x + threadIdx.x;
  if (idx < Nc * Nc) {
    int n = idx >> 9, m = idx & 511;
    sup[n * 1024 + m]       = (__bf16)s0[idx];
    sup[n * 1024 + 512 + m] = (__bf16)s1[idx];
    return;
  }
  int i2 = idx - Nc * Nc;
  if (i2 < 2048) {
    int o = i2 >> 6, f = i2 & 63, k = f >> 5, c = f & 31;
    wconv[i2]        = (__bf16)fw[o * 64 + c * 2 + k];  // filt_w (O,C,K)
    wconv[2048 + i2] = (__bf16)gw[o * 64 + c * 2 + k];  // gate_w
    return;
  }
  int i3 = i2 - 2048;
  if (i3 < 1024) {
    gcw[i3]        = (__bf16)g0w[i3];
    gcw[1024 + i3] = (__bf16)g1w[i3];
    swb[i3]        = (__bf16)skw[i3];
    rwb[i3]        = (__bf16)rsw[i3];
    return;
  }
  int i4 = i3 - 1024;
  if (i4 < 64) bn_acc[i4] = 0.f;
}

// ---------------- k1: causal conv (WMMA) + gated act + channel-mix into U ----------------
// one workgroup per (b,n); 128 threads = 4 waves
__global__ void k1_conv_u(const float* __restrict__ x,
                          const float* __restrict__ filt_b,
                          const float* __restrict__ gate_b,
                          const __bf16* __restrict__ wconv, // [2][32][64]
                          const __bf16* __restrict__ gcw,   // [2][32][32]
                          __bf16* __restrict__ u)           // [B][1024][4096]
{
  // Conv B operand (64 x 128) stored fragment-ready: 2 K-chunks of 32, each swizzled.
  // chunk kc, within-chunk row kk, col l: xb[kc*4096 + bswz(kk, l)]
  // chunk0 (k=0 tap) holds x[c][l-2] -> value x[c][p] written at col p+2; cols 0,1 zero.
  // chunk1 (k=1 tap) holds x[c][l].
  __shared__ __attribute__((aligned(32))) __bf16 xb[2 * 32 * 128];
  __shared__ __attribute__((aligned(32))) __bf16 wf[32 * 64], wgt[32 * 64];
  __shared__ __attribute__((aligned(32))) __bf16 gw0[32 * 32], gw1[32 * 32];
  __shared__ __attribute__((aligned(32))) __bf16 hs[32 * 128]; // swizzled (bswz)
  __shared__ float fb[32], gb[32];
  int wgid = blockIdx.x;
  int b = wgid >> 9, n = wgid & 511;
  int tid = threadIdx.x;

  for (int i = tid; i < Cc * Lc; i += 128) {
    int c = i >> 7, l = i & 127;
    __bf16 v = (__bf16)x[(((size_t)b * Cc + c) * Nc + n) * Lc + l];
    xb[4096 + bswz(c, l)] = v;                 // k=1 tap
    if (l + 2 < Lc) xb[bswz(c, l + 2)] = v;    // k=0 tap (causal shift)
  }
  if (tid < 64) {                               // zero causal pad: chunk0 cols 0,1
    int c = tid >> 1, col = tid & 1;
    xb[bswz(c, col)] = (__bf16)0.f;
  }
  for (int i = tid; i < 2048; i += 128) { wf[i] = wconv[i]; wgt[i] = wconv[2048 + i]; }
  for (int i = tid; i < 1024; i += 128) { gw0[i] = gcw[i]; gw1[i] = gcw[1024 + i]; }
  if (tid < 32) { fb[tid] = filt_b[tid]; gb[tid] = gate_b[tid]; }
  __syncthreads();

  int wave = tid >> 5, lane = tid & 31, half = lane >> 4;

  // conv GEMM: [filt;gate](32x128) = W(32x64) @ X(64x128); 2x8 tiles, 4 per wave
  for (int t = wave * 4; t < wave * 4 + 4; ++t) {
    int mo = t >> 3, lt = t & 7;
    v8f accf = {}; v8f accg = {};
#pragma unroll
    for (int kc = 0; kc < 2; ++kc) {
      v16bf bf = load_bfrag_sw(xb + kc * 4096, lt);
      v16bf af = load_afrag(wf,  64, mo * 16, kc * 32);
      accf = WMMA_BF16(af, bf, accf);
      v16bf ag = load_afrag(wgt, 64, mo * 16, kc * 32);
      accg = WMMA_BF16(ag, bf, accg);
    }
    int col = lt * 16 + (lane & 15);
#pragma unroll
    for (int r = 0; r < 8; ++r) {
      int o = mo * 16 + r + 8 * half;
      float h = fast_tanh(accf[r] + fb[o]) * fast_sigmoid(accg[r] + gb[o]);
      hs[bswz(o, col)] = (__bf16)h;   // store fragment-ready for stage C
    }
  }
  __syncthreads();

  // channel pre-mix: u0 = gc_w0 @ h, u1 = gc_w1 @ h (K=32, one wmma step)
  size_t ubase = (size_t)b * 1024 * 4096;
  for (int tt = wave * 8; tt < wave * 8 + 8; ++tt) {
    int s = tt >> 4, mo = (tt >> 3) & 1, lt = tt & 7;
    v16bf bfrag = load_bfrag_sw(hs, lt);
    v16bf afrag = load_afrag(s ? gw1 : gw0, 32, mo * 16, 0);
    v8f acc = {};
    acc = WMMA_BF16(afrag, bfrag, acc);
    int col = lt * 16 + (lane & 15);
    size_t rowbase = ubase + (size_t)(s * 512 + n) * 4096;
#pragma unroll
    for (int r = 0; r < 8; ++r) {
      int o = mo * 16 + r + 8 * half;
      u[rowbase + o * 128 + col] = (__bf16)acc[r];
    }
  }
}

// ---------------- k2: fused node GEMM  hg[b] = [S0|S1](512x1024) @ U[b](1024x4096) ----------------
// grid (32 col-blocks, 4 row-blocks, 32 batches), 256 threads = 8 waves; 128x128 tile / WG
__global__ void k2_nodegemm(const __bf16* __restrict__ sup, // [512][1024]
                            const __bf16* __restrict__ u,   // [B][1024][4096]
                            const float*  __restrict__ gc_b,
                            __bf16* __restrict__ hg)        // [B][512][4096]
{
  __shared__ __attribute__((aligned(32))) uint32_t At_u[128 * 32 / 2]; // A: linear row-major
  __shared__ __attribute__((aligned(32))) __bf16   Bt[32 * 128];      // B: swizzled (bswz)
  const __bf16* At = (const __bf16*)At_u;

  int ct = blockIdx.x, rt = blockIdx.y, b = blockIdx.z;
  int tid = threadIdx.x, wave = tid >> 5, lane = tid & 31, half = lane >> 4;
  int rowblk = rt * 128, colblk = ct * 128;
  const uint32_t* Au = reinterpret_cast<const uint32_t*>(sup);
  const uint32_t* Bu = reinterpret_cast<const uint32_t*>(u + (size_t)b * 1024 * 4096);

  v8f acc[8];
#pragma unroll
  for (int t = 0; t < 8; ++t) acc[t] = (v8f){};

  for (int kk = 0; kk < 1024; kk += 32) {
    for (int i = tid; i < 2048; i += 256) {
      int r = i >> 4, kp = i & 15;                 // A: 128 rows x 16 dword-pairs
      At_u[i] = Au[(size_t)(rowblk + r) * 512 + (kk >> 1) + kp];
      int kr = i >> 6, cp = i & 63;                // B: 32 rows x 64 dword-pairs
      union { uint32_t u32; __bf16 h[2]; } cv;
      cv.u32 = Bu[(size_t)(kk + kr) * 2048 + (colblk >> 1) + cp];
      Bt[bswz(kr, 2 * cp)]     = cv.h[0];
      Bt[bswz(kr, 2 * cp + 1)] = cv.h[1];
    }
    __syncthreads();
    v16bf af = load_afrag(At, 32, wave * 16, 0);
#pragma unroll
    for (int t = 0; t < 8; ++t) {
      v16bf bf = load_bfrag_sw(Bt, t);
      acc[t] = WMMA_BF16(af, bf, acc[t]);
    }
    __syncthreads();
  }

  size_t base = (size_t)b * 512 * 4096;
  int rbase = rowblk + wave * 16 + 8 * half;
#pragma unroll
  for (int t = 0; t < 8; ++t) {
    int col = colblk + t * 16 + (lane & 15);
    float bias = gc_b[col >> 7];
#pragma unroll
    for (int r = 0; r < 8; ++r)
      hg[base + (size_t)(rbase + r) * 4096 + col] = (__bf16)(acc[t][r] + bias);
  }
}

// ---------------- k3: skip & res channel mixes (WMMA) + BN partial sums ----------------
__global__ void k3_skipres(const __bf16* __restrict__ hg, // [B][512][4096]
                           const __bf16* __restrict__ sw, // [32][32]
                           const __bf16* __restrict__ rw, // [32][32]
                           const float* __restrict__ skip_b,
                           const float* __restrict__ res_b,
                           float* __restrict__ skip_out,  // (B,O,N,L)
                           __bf16* __restrict__ res,      // [B][N][C][L]
                           float* __restrict__ bn_acc)    // [64]
{
  __shared__ __attribute__((aligned(32))) __bf16 hsb[32 * 128]; // swizzled (bswz)
  __shared__ __attribute__((aligned(32))) __bf16 sws[32 * 32], rws[32 * 32];
  __shared__ float rs[32 * 128];
  __shared__ float sb[32], rb2[32];
  int wgid = blockIdx.x;
  int b = wgid >> 9, n = wgid & 511;
  int tid = threadIdx.x;
  const uint32_t* Hu = reinterpret_cast<const uint32_t*>(hg + (size_t)wgid * 4096);
  for (int i = tid; i < 2048; i += 128) {      // 2048 dwords = 32x128 bf16
    union { uint32_t u32; __bf16 h[2]; } cv;
    cv.u32 = Hu[i];
    int k = i >> 6, c0 = (i & 63) * 2;
    hsb[bswz(k, c0)]     = cv.h[0];
    hsb[bswz(k, c0 + 1)] = cv.h[1];
  }
  for (int i = tid; i < 1024; i += 128) { sws[i] = sw[i]; rws[i] = rw[i]; }
  if (tid < 32) { sb[tid] = skip_b[tid]; rb2[tid] = res_b[tid]; }
  __syncthreads();

  int wave = tid >> 5, lane = tid & 31, half = lane >> 4;
  for (int t = wave * 4; t < wave * 4 + 4; ++t) {
    int mo = t >> 3, lt = t & 7;
    v16bf bf = load_bfrag_sw(hsb, lt);
    v16bf as = load_afrag(sws, 32, mo * 16, 0);
    v8f accs = {}; accs = WMMA_BF16(as, bf, accs);
    v16bf ar = load_afrag(rws, 32, mo * 16, 0);
    v8f accr = {}; accr = WMMA_BF16(ar, bf, accr);
    int col = lt * 16 + (lane & 15);
#pragma unroll
    for (int r = 0; r < 8; ++r) {
      int o = mo * 16 + r + 8 * half;
      skip_out[(((size_t)b * Oc + o) * Nc + n) * Lc + col] = accs[r] + sb[o];
      float rv = accr[r] + rb2[o];
      __bf16 rbv = (__bf16)rv;
      res[(size_t)wgid * 4096 + o * 128 + col] = rbv;
      rs[o * 128 + col] = (float)rbv;
    }
  }
  __syncthreads();
  if (tid < 32) {  // block-level BN reduction -> one atomic per channel
    float s = 0.f, s2 = 0.f;
    for (int l = 0; l < 128; ++l) { float v = rs[tid * 128 + l]; s += v; s2 += v * v; }
    atomicAdd(&bn_acc[tid], s);
    atomicAdd(&bn_acc[32 + tid], s2);
  }
}

// ---------------- k4: batch-norm + residual ----------------
__global__ void k4_bn(const float* __restrict__ x,
                      const __bf16* __restrict__ res,   // [B][N][C][L]
                      const float* __restrict__ bn_acc,
                      const float* __restrict__ gamma,
                      const float* __restrict__ beta,
                      float* __restrict__ out)          // (B,C,N,L)
{
  __shared__ float smean[32], srstd[32];
  if (threadIdx.x < 32) {
    const float inv_cnt = 1.f / 2097152.f;  // B*N*L
    float mean = bn_acc[threadIdx.x] * inv_cnt;
    float var  = bn_acc[32 + threadIdx.x] * inv_cnt - mean * mean;
    smean[threadIdx.x] = mean;
    srstd[threadIdx.x] = rsqrtf(var + 1e-5f);
  }
  __syncthreads();
  size_t idx = (size_t)blockIdx.x * blockDim.x + threadIdx.x; // (B,C,N,L) flat, 2^26
  int l = idx & 127;
  int n = (int)((idx >> 7) & 511);
  int c = (int)((idx >> 16) & 31);
  int b = (int)(idx >> 21);
  float rv = (float)res[(((size_t)b * Nc + n) * Cc + c) * Lc + l];
  float y  = gamma[c] * (rv - smean[c]) * srstd[c] + beta[c];
  out[idx] = x[idx] + y;
}

// ---------------- launch ----------------
extern "C" void kernel_launch(void* const* d_in, const int* in_sizes, int n_in,
                              void* d_out, int out_size, void* d_ws, size_t ws_size,
                              hipStream_t stream) {
  (void)in_sizes; (void)n_in; (void)out_size; (void)ws_size;
  const float* x       = (const float*)d_in[0];
  const float* s0      = (const float*)d_in[1];
  const float* s1      = (const float*)d_in[2];
  const float* filt_w  = (const float*)d_in[3];
  const float* filt_b  = (const float*)d_in[4];
  const float* gate_w  = (const float*)d_in[5];
  const float* gate_b  = (const float*)d_in[6];
  const float* gc_w0   = (const float*)d_in[7];
  const float* gc_w1   = (const float*)d_in[8];
  const float* gc_b    = (const float*)d_in[9];
  const float* skip_w  = (const float*)d_in[10];
  const float* skip_b  = (const float*)d_in[11];
  const float* res_w   = (const float*)d_in[12];
  const float* res_b   = (const float*)d_in[13];
  const float* gamma   = (const float*)d_in[14];
  const float* beta    = (const float*)d_in[15];

  char* ws = (char*)d_ws;
  size_t off = 0;
  auto take = [&](size_t bytes) { char* p = ws + off; off = (off + bytes + 255) & ~(size_t)255; return p; };
  __bf16* sup    = (__bf16*)take((size_t)512 * 1024 * 2);      // 1 MB
  __bf16* wconv  = (__bf16*)take(4096 * 2);
  __bf16* gcw    = (__bf16*)take(2048 * 2);
  __bf16* swb    = (__bf16*)take(1024 * 2);
  __bf16* rwb    = (__bf16*)take(1024 * 2);
  float*  bn_acc = (float*) take(64 * 4);
  __bf16* u      = (__bf16*)take((size_t)32 * 1024 * 4096 * 2); // 256 MB
  __bf16* hg     = (__bf16*)take((size_t)32 * 512 * 4096 * 2);  // 128 MB
  __bf16* res    = (__bf16*)take((size_t)32 * 512 * 4096 * 2);  // 128 MB

  float* out_res  = (float*)d_out;                              // (B,C,N,L)
  float* out_skip = (float*)d_out + (size_t)Bc * Cc * Nc * Lc;  // (B,O,N,L)

  {
    int total = 512 * 512 + 2048 + 1024 + 64;
    k0_prep<<<(total + 255) / 256, 256, 0, stream>>>(s0, s1, filt_w, gate_w, gc_w0, gc_w1,
                                                     skip_w, res_w, sup, wconv, gcw, swb, rwb, bn_acc);
  }
  k1_conv_u<<<dim3(Bc * Nc), 128, 0, stream>>>(x, filt_b, gate_b, wconv, gcw, u);
  k2_nodegemm<<<dim3(32, 4, 32), 256, 0, stream>>>(sup, u, gc_b, hg);
  k3_skipres<<<dim3(Bc * Nc), 128, 0, stream>>>(hg, swb, rwb, skip_b, res_b, out_skip, res, bn_acc);
  k4_bn<<<dim3((unsigned)(((size_t)Bc * Cc * Nc * Lc) / 256)), 256, 0, stream>>>(x, res, bn_acc, gamma, beta, out_res);
}
